// TripleInputDirectDFF_21569325760540
// MI455X (gfx1250) — matrix operand order repaired and checked by reading, not compiled, not run on hardware
//
#include <hip/hip_runtime.h>
#include <stdint.h>

#define D_   768
#define D3_  2304
#define L_   512
#define N_   32
#define T_   (L_ * N_)   /* 16384 tokens */

#define USE_ASYNC_COPY 1

typedef unsigned short u16;
typedef __attribute__((ext_vector_type(16))) __bf16 v16bf;
typedef __attribute__((ext_vector_type(8)))  float  v8f;

union FragBF { uint4 q[2]; v16bf v; };

__device__ __forceinline__ u16 f2bf(float f) {
  union { float f; unsigned u; } x; x.f = f;
  unsigned u = x.u;
  u += 0x7FFFu + ((u >> 16) & 1u);   // round-to-nearest-even
  return (u16)(u >> 16);
}
__device__ __forceinline__ float bf2f(u16 h) {
  union { unsigned u; float f; } x; x.u = ((unsigned)h) << 16;
  return x.f;
}
__device__ __forceinline__ float frcp_(float x) { return __builtin_amdgcn_rcpf(x); }
__device__ __forceinline__ float sigmoidf_(float x) { return frcp_(1.0f + __expf(-x)); }
__device__ __forceinline__ float tanhf_(float x) {
  float cx = fminf(fmaxf(x, -15.f), 15.f);
  float e = __expf(2.0f * cx);
  return (e - 1.0f) * frcp_(e + 1.0f);
}

// async global->LDS copy of 16B per lane (gfx1250 GLOBAL_LOAD_ASYNC_TO_LDS_B128)
__device__ __forceinline__ void async_b128(const void* sbase, unsigned gbyteoff, void* ldsdst) {
  unsigned lds = (unsigned)(size_t)ldsdst;   // low 32 bits of shared flat addr = LDS byte addr
  asm volatile("global_load_async_to_lds_b128 %0, %1, %2"
               :: "v"(lds), "v"(gbyteoff), "s"(sbase) : "memory");
}

// ---------------------------------------------------------------------------
// Generic bf16 WMMA GEMM: C = A(bf16, lda) @ B(bf16, K x Nout row-major)
// EPI 0: store fp32.  EPI 1: +bias, ReLU, store bf16.
// Block 256 thr = 8 waves; block tile 64x256; wave tile 32x64; K-step 32.
// ---------------------------------------------------------------------------
template<int EPI>
__global__ __launch_bounds__(256) void gemm_bf16(
    const u16* __restrict__ A, int lda,
    const u16* __restrict__ B, int Nout, int K,
    const float* __restrict__ bias,
    float* __restrict__ Cf, u16* __restrict__ Cb)
{
  __shared__ alignas(16) u16 As[64][40];         // row pad -> 80B rows (16B mult)
  __shared__ alignas(16) u16 Bs[16][32][16];     // fragment-swizzled, 16 N-subtiles

  const int tid  = threadIdx.x;
  const int lane = tid & 31;
  const int wave = tid >> 5;
  const int wm   = wave >> 2;       // 0..1 (rows of 32)
  const int wn   = wave & 3;        // 0..3 (cols of 64)
  const int hlf  = lane >> 4;
  const int mrow = lane & 15;

  const int blockM = blockIdx.y * 64;
  const int blockN = blockIdx.x * 256;

  v8f acc[2][4];
  #pragma unroll
  for (int sm = 0; sm < 2; ++sm)
    #pragma unroll
    for (int jn = 0; jn < 4; ++jn)
      #pragma unroll
      for (int r = 0; r < 8; ++r) acc[sm][jn][r] = 0.0f;

  const int aRow = tid >> 2;            // 0..63
  const int aCol = (tid & 3) * 8;       // 0,8,16,24
  const int bRow = tid >> 5;            // 0..7
  const int bCol = (tid & 31) * 8;      // 0..248

  for (int k0 = 0; k0 < K; k0 += 32) {
    // stage A tile (64x32) row-major: async global->LDS, 16B per lane
#if USE_ASYNC_COPY
    async_b128(A, (unsigned)((((size_t)(blockM + aRow) * lda) + (k0 + aCol)) * sizeof(u16)),
               &As[aRow][aCol]);
#else
    uint4 av = *(const uint4*)(A + (size_t)(blockM + aRow) * lda + (k0 + aCol));
    *(uint4*)&As[aRow][aCol] = av;
#endif
    // stage B tile (32x256) swizzled into fragment order
    #pragma unroll
    for (int kk2 = 0; kk2 < 4; ++kk2) {
      int kr = bRow + kk2 * 8;
      uint4 bv = *(const uint4*)(B + (size_t)(k0 + kr) * Nout + (blockN + bCol));
      union { uint4 q; u16 s[8]; } u; u.q = bv;
      int bh = kr >> 4, kk = kr & 15;
      #pragma unroll
      for (int e = 0; e < 8; ++e) {
        int col = bCol + e;
        Bs[col >> 4][bh * 16 + (col & 15)][kk] = u.s[e];
      }
    }
    if (k0 + 32 < K)
      __builtin_prefetch(B + (size_t)(k0 + 32 + bRow) * Nout + (blockN + bCol), 0, 0);
#if USE_ASYNC_COPY
    __builtin_amdgcn_s_wait_asynccnt(0);
#endif
    __syncthreads();

    FragBF afr[2], bfr[4];
    #pragma unroll
    for (int sm = 0; sm < 2; ++sm) {
      const u16* ap = &As[wm * 32 + sm * 16 + mrow][0];
      afr[sm].q[0] = *(const uint4*)(ap + 16 * hlf);
      afr[sm].q[1] = *(const uint4*)(ap + 16 * hlf + 8);
    }
    #pragma unroll
    for (int jn = 0; jn < 4; ++jn) {
      const uint4* bp = (const uint4*)&Bs[wn * 4 + jn][lane][0];
      bfr[jn].q[0] = bp[0]; bfr[jn].q[1] = bp[1];
    }
    #pragma unroll
    for (int sm = 0; sm < 2; ++sm)
      #pragma unroll
      for (int jn = 0; jn < 4; ++jn)
        acc[sm][jn] = __builtin_amdgcn_wmma_f32_16x16x32_bf16(
            false, afr[sm].v, false, bfr[jn].v, (short)0, acc[sm][jn], false, false);
    __syncthreads();
  }

  #pragma unroll
  for (int sm = 0; sm < 2; ++sm)
    #pragma unroll
    for (int jn = 0; jn < 4; ++jn)
      #pragma unroll
      for (int r = 0; r < 8; ++r) {
        int row = blockM + wm * 32 + sm * 16 + r + 8 * hlf;
        int col = blockN + (wn * 4 + jn) * 16 + mrow;
        float v = acc[sm][jn][r];
        if (EPI == 0) {
          Cf[(size_t)row * Nout + col] = v;
        } else {
          v += bias[col];
          v = v > 0.f ? v : 0.f;
          Cb[(size_t)row * Nout + col] = f2bf(v);
        }
      }
}

// ---------------------------------------------------------------------------
// Fused gate/genh GEMMs + msa/enhanced-attention epilogue:
//   wce = wc * msa * sigmoid(wc@Wg) * (1 + 0.1*tanh(wc@Wh + genh_b))
// att layout: [7][32][2304] rows: 0 = a1, 1..2 = a2, 3..6 = a4
// Block tile 64x128; wave tile 32x32; dual accumulators.
// ---------------------------------------------------------------------------
__global__ __launch_bounds__(256) void gemm_gate_genh(
    const u16* __restrict__ wc,
    const u16* __restrict__ Wg, const u16* __restrict__ Wh,
    const float* __restrict__ genh_b,
    const float* __restrict__ att,
    u16* __restrict__ wce)
{
  __shared__ alignas(16) u16 As[64][40];
  __shared__ alignas(16) u16 Bg[8][32][16];
  __shared__ alignas(16) u16 Bh[8][32][16];

  const int tid  = threadIdx.x;
  const int lane = tid & 31;
  const int wave = tid >> 5;
  const int wm   = wave >> 2;
  const int wn   = wave & 3;
  const int hlf  = lane >> 4;
  const int mrow = lane & 15;

  const int blockM = blockIdx.y * 64;
  const int blockN = blockIdx.x * 128;

  v8f accG[2][2], accH[2][2];
  #pragma unroll
  for (int sm = 0; sm < 2; ++sm)
    #pragma unroll
    for (int jn = 0; jn < 2; ++jn)
      #pragma unroll
      for (int r = 0; r < 8; ++r) { accG[sm][jn][r] = 0.0f; accH[sm][jn][r] = 0.0f; }

  const int aRow = tid >> 2;
  const int aCol = (tid & 3) * 8;
  const int bRow = tid >> 4;
  const int bCol = (tid & 15) * 8;

  for (int k0 = 0; k0 < D3_; k0 += 32) {
#if USE_ASYNC_COPY
    async_b128(wc, (unsigned)((((size_t)(blockM + aRow) * D3_) + (k0 + aCol)) * sizeof(u16)),
               &As[aRow][aCol]);
#else
    uint4 av = *(const uint4*)(wc + (size_t)(blockM + aRow) * D3_ + (k0 + aCol));
    *(uint4*)&As[aRow][aCol] = av;
#endif
    #pragma unroll
    for (int kk2 = 0; kk2 < 2; ++kk2) {
      int kr = bRow + kk2 * 16;
      uint4 gv = *(const uint4*)(Wg + (size_t)(k0 + kr) * D3_ + (blockN + bCol));
      uint4 hv = *(const uint4*)(Wh + (size_t)(k0 + kr) * D3_ + (blockN + bCol));
      union { uint4 q; u16 s[8]; } ug, uh; ug.q = gv; uh.q = hv;
      int bh = kr >> 4, kk = kr & 15;
      #pragma unroll
      for (int e = 0; e < 8; ++e) {
        int col = bCol + e;
        int j = col >> 4, Ld = bh * 16 + (col & 15);
        Bg[j][Ld][kk] = ug.s[e];
        Bh[j][Ld][kk] = uh.s[e];
      }
    }
#if USE_ASYNC_COPY
    __builtin_amdgcn_s_wait_asynccnt(0);
#endif
    __syncthreads();

    FragBF afr[2], gfr[2], hfr[2];
    #pragma unroll
    for (int sm = 0; sm < 2; ++sm) {
      const u16* ap = &As[wm * 32 + sm * 16 + mrow][0];
      afr[sm].q[0] = *(const uint4*)(ap + 16 * hlf);
      afr[sm].q[1] = *(const uint4*)(ap + 16 * hlf + 8);
    }
    #pragma unroll
    for (int jn = 0; jn < 2; ++jn) {
      const uint4* gp = (const uint4*)&Bg[wn * 2 + jn][lane][0];
      const uint4* hp = (const uint4*)&Bh[wn * 2 + jn][lane][0];
      gfr[jn].q[0] = gp[0]; gfr[jn].q[1] = gp[1];
      hfr[jn].q[0] = hp[0]; hfr[jn].q[1] = hp[1];
    }
    #pragma unroll
    for (int sm = 0; sm < 2; ++sm)
      #pragma unroll
      for (int jn = 0; jn < 2; ++jn) {
        accG[sm][jn] = __builtin_amdgcn_wmma_f32_16x16x32_bf16(
            false, afr[sm].v, false, gfr[jn].v, (short)0, accG[sm][jn], false, false);
        accH[sm][jn] = __builtin_amdgcn_wmma_f32_16x16x32_bf16(
            false, afr[sm].v, false, hfr[jn].v, (short)0, accH[sm][jn], false, false);
      }
    __syncthreads();
  }

  #pragma unroll
  for (int sm = 0; sm < 2; ++sm)
    #pragma unroll
    for (int jn = 0; jn < 2; ++jn)
      #pragma unroll
      for (int r = 0; r < 8; ++r) {
        int t = blockM + wm * 32 + sm * 16 + r + 8 * hlf;
        int c = blockN + (wn * 2 + jn) * 16 + mrow;
        int l = t >> 5, n = t & 31;
        // align_corners interpolation weights
        float f2 = (float)l * (1.0f / 511.0f);
        float p4 = (float)l * (3.0f / 511.0f);
        int   i0 = (int)p4; i0 = i0 > 2 ? 2 : i0;
        float f4 = p4 - (float)i0;
        float a1v = att[(size_t)(0 * N_ + n) * D3_ + c];
        float a2v = (1.f - f2) * att[(size_t)((1 + 0) * N_ + n) * D3_ + c]
                  +        f2  * att[(size_t)((1 + 1) * N_ + n) * D3_ + c];
        float a4v = (1.f - f4) * att[(size_t)((3 + i0) * N_ + n) * D3_ + c]
                  +        f4  * att[(size_t)((3 + i0 + 1) * N_ + n) * D3_ + c];
        float msa = 0.5f * a1v + 0.3f * a2v + 0.2f * a4v;
        float gate = sigmoidf_(accG[sm][jn][r]);
        float genh = tanhf_(accH[sm][jn][r] + genh_b[c]);
        float enh = msa * gate * (1.0f + 0.1f * genh);
        float wcv = bf2f(wc[(size_t)t * D3_ + c]);
        wce[(size_t)t * D3_ + c] = f2bf(wcv * enh);
      }
}

// ---------------------------------------------------------------------------
// elementwise / small kernels
// ---------------------------------------------------------------------------
__global__ void concat_kernel(const float* __restrict__ x1, const float* __restrict__ x2,
                              const float* __restrict__ x3, u16* __restrict__ Cc)
{
  size_t idx = (size_t)blockIdx.x * blockDim.x + threadIdx.x;
  if (idx >= (size_t)T_ * D3_) return;
  int c = (int)(idx % D3_);
  size_t t = idx / D3_;
  float v = (c < D_) ? x1[t * D_ + c] : (c < 2 * D_) ? x2[t * D_ + c - D_] : x3[t * D_ + c - 2 * D_];
  Cc[idx] = f2bf(v);
}

__global__ void f2bf_kernel(const float* __restrict__ in, u16* __restrict__ out, int n)
{
  int i = blockIdx.x * blockDim.x + threadIdx.x;
  if (i < n) out[i] = f2bf(in[i]);
}

__global__ void iw2_softmax_kernel(const u16* __restrict__ h1, const float* __restrict__ W2,
                                   const float* __restrict__ b2, float* __restrict__ w)
{
  int t = blockIdx.x * blockDim.x + threadIdx.x;
  if (t >= T_) return;
  float a0 = b2[0], a1 = b2[1], a2 = b2[2];
  const u16* hp = h1 + (size_t)t * D_;
  for (int k = 0; k < D_; ++k) {
    float h = bf2f(hp[k]);
    a0 += h * W2[k * 3 + 0]; a1 += h * W2[k * 3 + 1]; a2 += h * W2[k * 3 + 2];
  }
  float m = fmaxf(a0, fmaxf(a1, a2));
  float e0 = __expf(a0 - m), e1 = __expf(a1 - m), e2 = __expf(a2 - m);
  float inv = frcp_(e0 + e1 + e2);
  w[t * 3 + 0] = e0 * inv; w[t * 3 + 1] = e1 * inv; w[t * 3 + 2] = e2 * inv;
}

__global__ void wconcat_kernel(const u16* __restrict__ Cc, const float* __restrict__ w,
                               u16* __restrict__ wc)
{
  size_t idx = (size_t)blockIdx.x * blockDim.x + threadIdx.x;
  if (idx >= (size_t)T_ * D3_) return;
  int c = (int)(idx % D3_);
  size_t t = idx / D3_;
  wc[idx] = f2bf(bf2f(Cc[idx]) * w[t * 3 + c / D_]);
}

__global__ void pool4_kernel(const u16* __restrict__ wc, float* __restrict__ p4)
{
  int idx = blockIdx.x * blockDim.x + threadIdx.x;
  if (idx >= 4 * N_ * D3_) return;
  int c = idx % D3_, n = (idx / D3_) % N_, chunk = idx / (D3_ * N_);
  float s = 0.f;
  for (int l = 0; l < 128; ++l)
    s += bf2f(wc[(size_t)((chunk * 128 + l) * N_ + n) * D3_ + c]);
  p4[idx] = s * (1.0f / 128.0f);
}

__global__ void pool_derive_kernel(float* __restrict__ pooled)
{
  int idx = blockIdx.x * blockDim.x + threadIdx.x;
  if (idx >= N_ * D3_) return;
  const float* p4 = pooled + (size_t)3 * N_ * D3_;
  float a = p4[idx], b = p4[(size_t)N_ * D3_ + idx];
  float c = p4[(size_t)2 * N_ * D3_ + idx], d = p4[(size_t)3 * N_ * D3_ + idx];
  pooled[(size_t)1 * N_ * D3_ + idx] = 0.5f * (a + b);   // p2[0]
  pooled[(size_t)2 * N_ * D3_ + idx] = 0.5f * (c + d);   // p2[1]
  pooled[idx] = 0.25f * (a + b + c + d);                 // p1
}

// one block per pooled row: out = sigmoid(LN(p @ W, g, b)), width 2304
__global__ __launch_bounds__(256) void att_row_kernel(
    const float* __restrict__ P, const float* __restrict__ W,
    const float* __restrict__ g, const float* __restrict__ b,
    float* __restrict__ out)
{
  __shared__ float prow[D3_];
  __shared__ float r1s[256];
  __shared__ float r2s[256];
  int row = blockIdx.x, tid = threadIdx.x;
  for (int c = tid; c < D3_; c += 256) prow[c] = P[(size_t)row * D3_ + c];
  __syncthreads();
  float vals[9];
  #pragma unroll
  for (int i = 0; i < 9; ++i) vals[i] = 0.f;
  for (int k = 0; k < D3_; ++k) {
    float pk = prow[k];
    #pragma unroll
    for (int i = 0; i < 9; ++i) vals[i] += pk * W[(size_t)k * D3_ + tid + i * 256];
  }
  float s = 0.f, q = 0.f;
  #pragma unroll
  for (int i = 0; i < 9; ++i) { s += vals[i]; q += vals[i] * vals[i]; }
  r1s[tid] = s; r2s[tid] = q; __syncthreads();
  for (int st = 128; st > 0; st >>= 1) {
    if (tid < st) { r1s[tid] += r1s[tid + st]; r2s[tid] += r2s[tid + st]; }
    __syncthreads();
  }
  float mean = r1s[0] * (1.0f / D3_);
  float var  = r2s[0] * (1.0f / D3_) - mean * mean;
  float inv  = rsqrtf(var + 1e-5f);
  #pragma unroll
  for (int i = 0; i < 9; ++i) {
    int c = tid + i * 256;
    float y = (vals[i] - mean) * inv * g[c] + b[c];
    out[(size_t)row * D3_ + c] = sigmoidf_(y);
  }
}

// LN over 768 cols. mode 0: relu -> fp32 outF. mode 1: bf16 -> outB[row*outLd+colOff+c].
// mode 2: add into fp32 outF.
__global__ __launch_bounds__(256) void ln768_kernel(
    const float* __restrict__ X, const float* __restrict__ gg, const float* __restrict__ bb,
    float* __restrict__ outF, u16* __restrict__ outB, int outLd, int colOff, int mode)
{
  __shared__ float r1s[256];
  __shared__ float r2s[256];
  int row = blockIdx.x, tid = threadIdx.x;
  float v[3];
  #pragma unroll
  for (int i = 0; i < 3; ++i) v[i] = X[(size_t)row * D_ + tid + i * 256];
  r1s[tid] = v[0] + v[1] + v[2];
  r2s[tid] = v[0] * v[0] + v[1] * v[1] + v[2] * v[2];
  __syncthreads();
  for (int st = 128; st > 0; st >>= 1) {
    if (tid < st) { r1s[tid] += r1s[tid + st]; r2s[tid] += r2s[tid + st]; }
    __syncthreads();
  }
  float mean = r1s[0] * (1.0f / D_);
  float var  = r2s[0] * (1.0f / D_) - mean * mean;
  float inv  = rsqrtf(var + 1e-5f);
  #pragma unroll
  for (int i = 0; i < 3; ++i) {
    int c = tid + i * 256;
    float y = (v[i] - mean) * inv * gg[c] + bb[c];
    if (mode == 0)      outF[(size_t)row * D_ + c] = fmaxf(y, 0.f);
    else if (mode == 1) outB[(size_t)row * outLd + colOff + c] = f2bf(y);
    else                outF[(size_t)row * D_ + c] += y;
  }
}

__global__ void finalatt_kernel(
    const float* __restrict__ x1, const float* __restrict__ x2, const float* __restrict__ x3,
    const float* __restrict__ fc1W, const float* __restrict__ fc1b,
    const float* __restrict__ fc2W, const float* __restrict__ fc2b,
    const float* __restrict__ fc3W, const float* __restrict__ fc3b,
    const float* __restrict__ afW, const float* __restrict__ afb,
    float* __restrict__ fa)
{
  int t = blockIdx.x * blockDim.x + threadIdx.x;
  if (t >= T_) return;
  float s1 = fc1b[0], s2 = fc2b[0], s3 = fc3b[0];
  const float* p1 = x1 + (size_t)t * D_;
  const float* p2 = x2 + (size_t)t * D_;
  const float* p3 = x3 + (size_t)t * D_;
  for (int k = 0; k < D_; ++k) {
    s1 += p1[k] * fc1W[k];
    s2 += p2[k] * fc2W[k];
    s3 += p3[k] * fc3W[k];
  }
  float z = s1 * afW[0] + s2 * afW[1] + s3 * afW[2] + afb[0];
  fa[t] = sigmoidf_(z);
}

__global__ void scale_kernel(float* __restrict__ out, const float* __restrict__ fa)
{
  size_t idx = (size_t)blockIdx.x * blockDim.x + threadIdx.x;
  if (idx >= (size_t)T_ * D_) return;
  out[idx] *= fa[idx / D_];
}

// ---------------------------------------------------------------------------
extern "C" void kernel_launch(void* const* d_in, const int* in_sizes, int n_in,
                              void* d_out, int out_size, void* d_ws, size_t ws_size,
                              hipStream_t stream)
{
  (void)in_sizes; (void)n_in; (void)out_size; (void)ws_size;
  const float* input1 = (const float*)d_in[0];
  const float* input2 = (const float*)d_in[1];
  const float* input3 = (const float*)d_in[2];
  const float* iw_W1  = (const float*)d_in[3];
  const float* iw_b1  = (const float*)d_in[4];
  const float* iw_W2  = (const float*)d_in[5];
  const float* iw_b2  = (const float*)d_in[6];
  const float* att1_W = (const float*)d_in[7];
  const float* att1_g = (const float*)d_in[8];
  const float* att1_b = (const float*)d_in[9];
  const float* att2_W = (const float*)d_in[10];
  const float* att2_g = (const float*)d_in[11];
  const float* att2_b = (const float*)d_in[12];
  const float* att4_W = (const float*)d_in[13];
  const float* att4_g = (const float*)d_in[14];
  const float* att4_b = (const float*)d_in[15];
  const float* gate_W = (const float*)d_in[16];
  const float* genh_W = (const float*)d_in[17];
  const float* genh_b = (const float*)d_in[18];
  const float* redu_W = (const float*)d_in[19];
  const float* redu_g = (const float*)d_in[20];
  const float* redu_b = (const float*)d_in[21];
  const float* r1_W = (const float*)d_in[22];
  const float* r1_g = (const float*)d_in[23];
  const float* r1_b = (const float*)d_in[24];
  const float* r2_W = (const float*)d_in[25];
  const float* r2_g = (const float*)d_in[26];
  const float* r2_b = (const float*)d_in[27];
  const float* r3_W = (const float*)d_in[28];
  const float* r3_g = (const float*)d_in[29];
  const float* r3_b = (const float*)d_in[30];
  const float* rf_W = (const float*)d_in[31];
  const float* rf_g = (const float*)d_in[32];
  const float* rf_b = (const float*)d_in[33];
  const float* fc1_W = (const float*)d_in[34];
  const float* fc1_b = (const float*)d_in[35];
  const float* fc2_W = (const float*)d_in[36];
  const float* fc2_b = (const float*)d_in[37];
  const float* fc3_W = (const float*)d_in[38];
  const float* fc3_b = (const float*)d_in[39];
  const float* af_W  = (const float*)d_in[40];
  const float* af_b  = (const float*)d_in[41];

  char* ws = (char*)d_ws;
  size_t off = 0;
  auto take = [&](size_t bytes) { size_t o = off; off += (bytes + 255) & ~(size_t)255; return o; };

  const size_t SZ_BIG = (size_t)T_ * D3_ * 2;             // 75.5 MB bf16
  u16* Cc    = (u16*)(ws + take(SZ_BIG));                 // concat bf16
  u16* wcb   = (u16*)(ws + take(SZ_BIG));                 // wc bf16 (later: res)
  u16* wce   = (u16*)(ws + take(SZ_BIG));                 // wc * enhanced_att
  u16* iw1b  = (u16*)(ws + take((size_t)D3_ * D_ * 2));
  u16* gateb = (u16*)(ws + take((size_t)D3_ * D3_ * 2));
  u16* genhb = (u16*)(ws + take((size_t)D3_ * D3_ * 2));
  u16* redub = (u16*)(ws + take((size_t)D3_ * D_ * 2));
  u16* r1b   = (u16*)(ws + take((size_t)D_ * D_ * 2));
  u16* r2b   = (u16*)(ws + take((size_t)D_ * D_ * 2));
  u16* r3b   = (u16*)(ws + take((size_t)D_ * D_ * 2));
  u16* rfb   = (u16*)(ws + take((size_t)D3_ * D_ * 2));
  float* tmp = (float*)(ws + take((size_t)T_ * D_ * 4));  // fp32 pre-LN; h1 overlays
  u16* h1    = (u16*)tmp;                                 // T x 768 bf16 (early phase only)
  float* w3  = (float*)(ws + take((size_t)T_ * 3 * 4));
  float* pooled = (float*)(ws + take((size_t)7 * N_ * D3_ * 4));
  float* attb   = (float*)(ws + take((size_t)7 * N_ * D3_ * 4));
  float* fa     = (float*)(ws + take((size_t)T_ * 4));
  float* out    = (float*)d_out;

  const int EL = 256;
  const size_t nBig = (size_t)T_ * D3_;

  // 1) concat -> bf16
  concat_kernel<<<(unsigned)((nBig + EL - 1) / EL), EL, 0, stream>>>(input1, input2, input3, Cc);

  // 2) weight conversions to bf16
  f2bf_kernel<<<(D3_ * D_ + EL - 1) / EL, EL, 0, stream>>>(iw_W1, iw1b, D3_ * D_);
  f2bf_kernel<<<(D3_ * D3_ + EL - 1) / EL, EL, 0, stream>>>(gate_W, gateb, D3_ * D3_);
  f2bf_kernel<<<(D3_ * D3_ + EL - 1) / EL, EL, 0, stream>>>(genh_W, genhb, D3_ * D3_);
  f2bf_kernel<<<(D3_ * D_ + EL - 1) / EL, EL, 0, stream>>>(redu_W, redub, D3_ * D_);
  f2bf_kernel<<<(D_ * D_ + EL - 1) / EL, EL, 0, stream>>>(r1_W, r1b, D_ * D_);
  f2bf_kernel<<<(D_ * D_ + EL - 1) / EL, EL, 0, stream>>>(r2_W, r2b, D_ * D_);
  f2bf_kernel<<<(D_ * D_ + EL - 1) / EL, EL, 0, stream>>>(r3_W, r3b, D_ * D_);
  f2bf_kernel<<<(D3_ * D_ + EL - 1) / EL, EL, 0, stream>>>(rf_W, rfb, D3_ * D_);

  // 3) h1 = relu(concat @ iw_W1 + b1)   [WMMA, bf16 out]
  gemm_bf16<1><<<dim3(D_ / 256, T_ / 64), 256, 0, stream>>>(
      Cc, D3_, iw1b, D_, D3_, iw_b1, nullptr, h1);

  // 4) w = softmax(h1 @ iw_W2 + b2)
  iw2_softmax_kernel<<<(T_ + EL - 1) / EL, EL, 0, stream>>>(h1, iw_W2, iw_b2, w3);

  // 5) weighted concat (bf16)
  wconcat_kernel<<<(unsigned)((nBig + EL - 1) / EL), EL, 0, stream>>>(Cc, w3, wcb);

  // 6/7) adaptive pools k=4 then derive k=2,1
  pool4_kernel<<<(4 * N_ * D3_ + EL - 1) / EL, EL, 0, stream>>>(wcb, pooled + (size_t)3 * N_ * D3_);
  pool_derive_kernel<<<(N_ * D3_ + EL - 1) / EL, EL, 0, stream>>>(pooled);

  // 8) pooled attentions: sigmoid(LN(p @ W))
  att_row_kernel<<<32, 256, 0, stream>>>(pooled, att1_W, att1_g, att1_b, attb);
  att_row_kernel<<<64, 256, 0, stream>>>(pooled + (size_t)1 * N_ * D3_, att2_W, att2_g, att2_b,
                                         attb + (size_t)1 * N_ * D3_);
  att_row_kernel<<<128, 256, 0, stream>>>(pooled + (size_t)3 * N_ * D3_, att4_W, att4_g, att4_b,
                                          attb + (size_t)3 * N_ * D3_);

  // 9) fused gate/genh GEMMs + enhanced-attention epilogue -> wce (bf16)
  gemm_gate_genh<<<dim3(D3_ / 128, T_ / 64), 256, 0, stream>>>(
      wcb, gateb, genhb, genh_b, attb, wce);

  // 10) red = relu(LN(wce @ redu_W)) -> out
  gemm_bf16<0><<<dim3(D_ / 256, T_ / 64), 256, 0, stream>>>(
      wce, D3_, redub, D_, D3_, nullptr, tmp, nullptr);
  ln768_kernel<<<T_, 256, 0, stream>>>(tmp, redu_g, redu_b, out, nullptr, 0, 0, 0);

  // 11) res = concat(LN(x_j @ rj_W))  (bf16, reuses wc region)
  u16* res = wcb;
  const u16* rW[3] = { r1b, r2b, r3b };
  const float* rg[3] = { r1_g, r2_g, r3_g };
  const float* rb[3] = { r1_b, r2_b, r3_b };
  for (int j = 0; j < 3; ++j) {
    gemm_bf16<0><<<dim3(D_ / 256, T_ / 64), 256, 0, stream>>>(
        Cc + (size_t)j * D_, D3_, rW[j], D_, D_, nullptr, tmp, nullptr);
    ln768_kernel<<<T_, 256, 0, stream>>>(tmp, rg[j], rb[j], nullptr, res, D3_, j * D_, 1);
  }

  // 12) out += LN(res @ rf_W)
  gemm_bf16<0><<<dim3(D_ / 256, T_ / 64), 256, 0, stream>>>(
      res, D3_, rfb, D_, D3_, nullptr, tmp, nullptr);
  ln768_kernel<<<T_, 256, 0, stream>>>(tmp, rf_g, rf_b, out, nullptr, 0, 0, 2);

  // 13/14) final attention scaling
  finalatt_kernel<<<(T_ + EL - 1) / EL, EL, 0, stream>>>(
      input1, input2, input3, fc1_W, fc1_b, fc2_W, fc2_b, fc3_W, fc3_b, af_W, af_b, fa);
  scale_kernel<<<(unsigned)(((size_t)T_ * D_ + EL - 1) / EL), EL, 0, stream>>>(out, fa);
}
